// MakeMoreRNN_26834955665477
// MI455X (gfx1250) — matrix operand order repaired and checked by reading, not compiled, not run on hardware
//
#include <hip/hip_runtime.h>
#include <hip/hip_bf16.h>

#define VOCAB 27
#define EMB   32
#define HID   64
#define BATCH 4096
#define TLEN  256

#define ROWS  16         // batch rows per wave (WMMA M)
#define HPAD  (HID + 2)  // pad LDS rows to 66 halves (33 dwords) -> conflict-free

typedef __attribute__((ext_vector_type(16))) __bf16 v16bf;
typedef __attribute__((ext_vector_type(8)))  float  v8f;

struct U16x16 { unsigned short x[16]; };

// f32 -> bf16 round-to-nearest-even (init-time only)
__device__ __forceinline__ unsigned short f2bf(float f) {
    unsigned int u = __float_as_uint(f);
    u += 0x7fffu + ((u >> 16) & 1u);
    return (unsigned short)(u >> 16);
}

// Branch-free tanh: native V_TANH_F32 on gfx1250 (confirmed), with fallback.
__device__ __forceinline__ float fast_tanh(float x) {
#if __has_builtin(__builtin_amdgcn_tanhf)
    return __builtin_amdgcn_tanhf(x);
#else
    const float ax = __builtin_fabsf(x);
    const float e  = exp2f(ax * 2.8853900817779268f);
    const float r  = 1.0f - 2.0f / (e + 1.0f);
    return __builtin_copysignf(r, x);
#endif
}

// 16-bit A-matrix (16xK) VGPR layout (contiguous k = 16*half + 2v, matching B fragments): pair base K for VGPR v, lane half
__device__ __forceinline__ int a_koff(int v, int half) {
    return 2 * v + 16 * half;
}

__device__ __forceinline__ v8f wmma_bf16(v16bf a, v16bf b, v8f c) {
    return __builtin_amdgcn_wmma_f32_16x16x32_bf16(
        /*neg_a=*/false, a, /*neg_b=*/false, b,
        /*c_mod=*/(short)0, c, /*reuse_a=*/false, /*reuse_b=*/false);
}

__global__ __launch_bounds__(32)
void rnn_lm_fused(const int* __restrict__ X,
                  const float* __restrict__ emb_w,
                  const float* __restrict__ start,
                  const float* __restrict__ rnn_w,   // (HID, EMB+HID)
                  const float* __restrict__ rnn_b,   // (HID)
                  const float* __restrict__ lm_w,    // (VOCAB, HID)
                  const float* __restrict__ lm_b,    // (VOCAB)
                  float* __restrict__ out)           // (B, T, VOCAB)
{
    __shared__ unsigned short ebf[VOCAB * EMB];      // bf16 embedding table
    __shared__ unsigned short hlds[ROWS][HPAD];      // h staging (one wave/block)

    const int lane = threadIdx.x & 31;
    const int half = lane >> 4;   // 0: lanes 0-15, 1: lanes 16-31
    const int mr   = lane & 15;   // A-row / B-col / D-col index

    // ---- bf16 embedding table into LDS (once) ----
    for (int i = lane; i < VOCAB * EMB; i += 32)
        ebf[i] = f2bf(emb_w[i]);
    __syncthreads();              // single-wave workgroup: hw NOP, compiler fence

    // ---- preload all weights as register-resident WMMA B-fragments ----
    v16bf WxB[4];        // B[k=e][n=h]: xproj,  K=32,  4 N-tiles of HID
    v16bf WhB[2][4];     // B[k=h][n=h]: recur,  K=64 (2 chunks), 4 N-tiles
    v16bf LmB[2][2];     // B[k=h][n=v]: logits, K=64 (2 chunks), 2 N-tiles

    #pragma unroll
    for (int nt = 0; nt < 4; ++nt) {
        U16x16 u;
        const int h = 16 * nt + mr;
        #pragma unroll
        for (int j = 0; j < 8; ++j) {
            const int e = 2 * j + 16 * half;
            u.x[2 * j]     = f2bf(rnn_w[h * (EMB + HID) + e]);
            u.x[2 * j + 1] = f2bf(rnn_w[h * (EMB + HID) + e + 1]);
        }
        WxB[nt] = __builtin_bit_cast(v16bf, u);
    }
    #pragma unroll
    for (int c = 0; c < 2; ++c)
        #pragma unroll
        for (int nt = 0; nt < 4; ++nt) {
            U16x16 u;
            const int n = 16 * nt + mr;
            #pragma unroll
            for (int j = 0; j < 8; ++j) {
                const int k = 32 * c + 2 * j + 16 * half;
                u.x[2 * j]     = f2bf(rnn_w[n * (EMB + HID) + EMB + k]);
                u.x[2 * j + 1] = f2bf(rnn_w[n * (EMB + HID) + EMB + k + 1]);
            }
            WhB[c][nt] = __builtin_bit_cast(v16bf, u);
        }
    #pragma unroll
    for (int c = 0; c < 2; ++c)
        #pragma unroll
        for (int lt = 0; lt < 2; ++lt) {
            U16x16 u;
            const int v = 16 * lt + mr;
            #pragma unroll
            for (int j = 0; j < 8; ++j) {
                const int k = 32 * c + 2 * j + 16 * half;
                if (v < VOCAB) {
                    u.x[2 * j]     = f2bf(lm_w[v * HID + k]);
                    u.x[2 * j + 1] = f2bf(lm_w[v * HID + k + 1]);
                } else {
                    u.x[2 * j] = 0; u.x[2 * j + 1] = 0;
                }
            }
            LmB[c][lt] = __builtin_bit_cast(v16bf, u);
        }

    // ---- persistent bias accumulators in C/D layout (built once, no per-step
    //      splat: WMMA reads them as SRC2 and writes a different VDST) ----
    v8f biasC[4];   // rnn_b broadcast: value depends only on N = mr
    #pragma unroll
    for (int nt = 0; nt < 4; ++nt) {
        const float b = rnn_b[16 * nt + mr];
        #pragma unroll
        for (int i = 0; i < 8; ++i) biasC[nt][i] = b;
    }
    v8f biasV[2];   // lm_b broadcast (zero-padded past VOCAB)
    #pragma unroll
    for (int lt = 0; lt < 2; ++lt) {
        const int v = 16 * lt + mr;
        const float b = (v < VOCAB) ? lm_b[v] : 0.0f;
        #pragma unroll
        for (int i = 0; i < 8; ++i) biasV[lt][i] = b;
    }

    const int b0 = blockIdx.x * ROWS;

    // ---- initial hidden state A-fragments from broadcast `start` ----
    v16bf hA[2];
    #pragma unroll
    for (int c = 0; c < 2; ++c) {
        U16x16 u;
        #pragma unroll
        for (int v = 0; v < 8; ++v) {
            const int k = 32 * c + a_koff(v, half);
            u.x[2 * v]     = f2bf(start[k]);
            u.x[2 * v + 1] = f2bf(start[k + 1]);
        }
        hA[c] = __builtin_bit_cast(v16bf, u);
    }

    const int rowX = (b0 + mr) * TLEN;
    int tok = X[rowX];   // software-pipelined token fetch

    for (int t = 0; t < TLEN; ++t) {
        // prefetch next token (clamped index: branch-free)
        const int tn = t + 1 < TLEN ? t + 1 : TLEN - 1;
        const int tok_next = X[rowX + tn];

        // ---- embedding A fragment (16x32 bf16) from LDS table ----
        v16bf eA;
        {
            U16x16 u;
            const unsigned short* er = &ebf[tok * EMB];
            #pragma unroll
            for (int v = 0; v < 8; ++v) {
                const int k = a_koff(v, half);
                const unsigned int p = *(const unsigned int*)(&er[k]);
                u.x[2 * v]     = (unsigned short)(p & 0xffffu);
                u.x[2 * v + 1] = (unsigned short)(p >> 16);
            }
            eA = __builtin_bit_cast(v16bf, u);
        }

        // ---- pre-activation: b + emb@Wx^T + h@Wh^T  (12 WMMAs) ----
        v8f acc[4];
        #pragma unroll
        for (int nt = 0; nt < 4; ++nt) {
            v8f c0 = wmma_bf16(eA,    WxB[nt],    biasC[nt]);   // C=bias, D=new
            c0     = wmma_bf16(hA[0], WhB[0][nt], c0);
            c0     = wmma_bf16(hA[1], WhB[1][nt], c0);
            acc[nt] = c0;
        }

        // ---- tanh (V_TANH_F32), stage h: round-half-up + hi16 store
        //      (folds to ds_store_b16_d16_hi: 1 VALU/element) ----
        #pragma unroll
        for (int nt = 0; nt < 4; ++nt)
            #pragma unroll
            for (int r = 0; r < 8; ++r) {
                const float hv = fast_tanh(acc[nt][r]);
                const unsigned int b = __float_as_uint(hv) + 0x8000u;
                hlds[r + 8 * half][16 * nt + mr] = (unsigned short)(b >> 16);
            }
        __syncthreads();   // hw NOP (single wave), orders LDS store->load

        // ---- rebuild h as A fragments for next step & logits ----
        {
            const unsigned short* hr = &hlds[mr][0];
            #pragma unroll
            for (int c = 0; c < 2; ++c) {
                U16x16 u;
                #pragma unroll
                for (int v = 0; v < 8; ++v) {
                    const int k = 32 * c + a_koff(v, half);
                    const unsigned int p = *(const unsigned int*)(&hr[k]);
                    u.x[2 * v]     = (unsigned short)(p & 0xffffu);
                    u.x[2 * v + 1] = (unsigned short)(p >> 16);
                }
                hA[c] = __builtin_bit_cast(v16bf, u);
            }
        }

        // ---- logits: lm_b + h@lm_w^T  (4 WMMAs), stream out NT ----
        #pragma unroll
        for (int lt = 0; lt < 2; ++lt) {
            v8f lacc = wmma_bf16(hA[0], LmB[0][lt], biasV[lt]);
            lacc     = wmma_bf16(hA[1], LmB[1][lt], lacc);
            const int n = 16 * lt + mr;
            if (n < VOCAB) {
                #pragma unroll
                for (int r = 0; r < 8; ++r) {
                    const size_t row = (size_t)(b0 + r + 8 * half);
                    __builtin_nontemporal_store(
                        lacc[r], &out[(row * TLEN + (size_t)t) * VOCAB + n]);
                }
            }
        }

        tok = tok_next;
    }
}

extern "C" void kernel_launch(void* const* d_in, const int* in_sizes, int n_in,
                              void* d_out, int out_size, void* d_ws, size_t ws_size,
                              hipStream_t stream) {
    const int*   X     = (const int*)d_in[0];
    const float* emb_w = (const float*)d_in[1];
    const float* start = (const float*)d_in[2];
    const float* rnn_w = (const float*)d_in[3];
    const float* rnn_b = (const float*)d_in[4];
    const float* lm_w  = (const float*)d_in[5];
    const float* lm_b  = (const float*)d_in[6];
    float* out = (float*)d_out;

    dim3 grid(BATCH / ROWS);   // 256 blocks, one wave each
    dim3 block(32);
    rnn_lm_fused<<<grid, block, 0, stream>>>(X, emb_w, start, rnn_w, rnn_b,
                                             lm_w, lm_b, out);
}
